// LIIF_65506841198648
// MI455X (gfx1250) — matrix operand order repaired and checked by reading, not compiled, not run on hardware
//
#include <hip/hip_runtime.h>
#include <hip/hip_bf16.h>

typedef __attribute__((ext_vector_type(16))) _Float16 v16h;
typedef __attribute__((ext_vector_type(8)))  _Float16 v8h;
typedef __attribute__((ext_vector_type(8)))  float    v8f;

// ---------------- problem constants ----------------
#define CCH   64
#define HF    128
#define WF    128
#define HWSZ  (HF * WF)
#define QD    65536
#define HID   256
#define K0PAD 608          // 580 padded up to 19 * 32
#define KT0   19
#define KT1   8
#define XSTR  624          // 608+16 f16 pad (312 dw, 56 mod 64 -> conflict-free rows)
#define HSTR  280          // 256+24 f16 pad (140 dw, 12 mod 64 -> conflict-free rows)
#define MROWS 32           // 8 queries x 4 corners per workgroup

// packed-weight layout offsets in d_ws (f16 elements)
#define L0_ELEMS (K0PAD * HID)        // 155648
#define L_ELEMS  (HID * HID)          // 65536
#define BASE1 (L0_ELEMS)
#define BASE2 (BASE1 + L_ELEMS)
#define BASE3 (BASE2 + L_ELEMS)
#define BASE4 (BASE3 + L_ELEMS)      // w4 packed 256x16 (3 valid cols)
#define W4_ELEMS (HID * 16)          // 4096
#define TOTAL_W (BASE4 + W4_ELEMS)   // 356352  (~713 KB f16 in d_ws)

// B-fragment swizzle for V_WMMA_F32_16X16X32_F16:
// lane = nn + 16*(kk>=16); vgpr v=(kk&15)>>1 holds K pair (low half = even K)
__device__ __forceinline__ int b_frag_idx(int base, int KT, int k, int n) {
    int kt = k >> 5, kk = k & 31;
    int nt = n >> 4, nn = n & 15;
    int lane = nn + ((kk & 16) ? 16 : 0);
    int v    = (kk & 15) >> 1;
    int half = kk & 1;
    return base + (((nt * KT + kt) << 9) | (lane << 4) | (v << 1) | half);
}

// ---------------- weight repack kernel ----------------
__global__ void liif_prep_weights(const float* __restrict__ w0,
                                  const float* __restrict__ w1,
                                  const float* __restrict__ w2,
                                  const float* __restrict__ w3,
                                  const float* __restrict__ w4,
                                  _Float16* __restrict__ ws) {
    int gid = blockIdx.x * 256 + threadIdx.x;
    if (gid >= TOTAL_W) return;
    if (gid < L0_ELEMS) {
        int k = gid >> 8, n = gid & 255;                  // k < 608
        float v = (k < 580) ? w0[k * HID + n] : 0.0f;     // zero-pad K
        ws[b_frag_idx(0, KT0, k, n)] = (_Float16)v;
    } else if (gid < BASE2) {
        int l = gid - BASE1; int k = l >> 8, n = l & 255;
        ws[b_frag_idx(BASE1, KT1, k, n)] = (_Float16)w1[k * HID + n];
    } else if (gid < BASE3) {
        int l = gid - BASE2; int k = l >> 8, n = l & 255;
        ws[b_frag_idx(BASE2, KT1, k, n)] = (_Float16)w2[k * HID + n];
    } else if (gid < BASE4) {
        int l = gid - BASE3; int k = l >> 8, n = l & 255;
        ws[b_frag_idx(BASE3, KT1, k, n)] = (_Float16)w3[k * HID + n];
    } else {
        int l = gid - BASE4; int k = l >> 4, n = l & 15;  // 256 x 16, pad N
        float v = (n < 3) ? w4[k * 3 + n] : 0.0f;
        ws[b_frag_idx(BASE4, KT1, k, n)] = (_Float16)v;
    }
}

// One WMMA layer over M=32 rows: each wave owns 2 N-tiles x 2 M-tiles.
// A from LDS f16, B fragments pre-swizzled in global (each reused for 2 M-tiles),
// bias + ReLU fused, D written to LDS f16 for the next layer's A.
// All base pointers hoisted so loads lower to {sbase + lane-voffset + imm}.
__device__ __forceinline__ void wmma_layer(const _Float16* __restrict__ Asrc, int astride,
                                           int KT,
                                           const _Float16* __restrict__ Wblk,
                                           const float* __restrict__ bias,
                                           _Float16* __restrict__ Hdst, int hstride,
                                           int lane, int wid) {
    const int laneHi = lane >> 4;
    const int m      = lane & 15;
    const int nn     = lane & 15;
    const int nt0 = wid * 2, nt1 = wid * 2 + 1;

    // hoisted bases: per-wave-uniform block term + per-lane term
    const _Float16* w0p = Wblk + nt0 * KT * 512 + lane * 16;
    const _Float16* w1p = Wblk + nt1 * KT * 512 + lane * 16;
    const _Float16* a0p = Asrc + m * astride + laneHi * 16;         // M-tile 0
    const _Float16* a1p = a0p + 16 * astride;                       // M-tile 1

    v8f acc00 = {}, acc01 = {}, acc10 = {}, acc11 = {};
    for (int kt = 0; kt < KT; ++kt) {
        v8h a0lo = *(const v8h*)(a0p + kt * 32);
        v8h a0hi = *(const v8h*)(a0p + kt * 32 + 8);
        v8h a1lo = *(const v8h*)(a1p + kt * 32);
        v8h a1hi = *(const v8h*)(a1p + kt * 32 + 8);
        v16h a0 = __builtin_shufflevector(a0lo, a0hi, 0,1,2,3,4,5,6,7,8,9,10,11,12,13,14,15);
        v16h a1 = __builtin_shufflevector(a1lo, a1hi, 0,1,2,3,4,5,6,7,8,9,10,11,12,13,14,15);
        v16h b0 = *(const v16h*)(w0p + kt * 512);
        v16h b1 = *(const v16h*)(w1p + kt * 512);
        acc00 = __builtin_amdgcn_wmma_f32_16x16x32_f16(false, a0, false, b0, (short)0, acc00, false, false);
        acc01 = __builtin_amdgcn_wmma_f32_16x16x32_f16(false, a0, false, b1, (short)0, acc01, false, false);
        acc10 = __builtin_amdgcn_wmma_f32_16x16x32_f16(false, a1, false, b0, (short)0, acc10, false, false);
        acc11 = __builtin_amdgcn_wmma_f32_16x16x32_f16(false, a1, false, b1, (short)0, acc11, false, false);
    }
    float bi0 = bias[nt0 * 16 + nn];
    float bi1 = bias[nt1 * 16 + nn];
    _Float16* h0 = Hdst + laneHi * 8 * hstride + nt0 * 16 + nn;
    _Float16* h1 = Hdst + laneHi * 8 * hstride + nt1 * 16 + nn;
#pragma unroll
    for (int r = 0; r < 8; ++r) {
        float v;
        v = acc00[r] + bi0; v = v > 0.0f ? v : 0.0f; h0[r * hstride]                  = (_Float16)v;
        v = acc01[r] + bi1; v = v > 0.0f ? v : 0.0f; h1[r * hstride]                  = (_Float16)v;
        v = acc10[r] + bi0; v = v > 0.0f ? v : 0.0f; h0[(r + 16) * hstride]           = (_Float16)v;
        v = acc11[r] + bi1; v = v > 0.0f ? v : 0.0f; h1[(r + 16) * hstride]           = (_Float16)v;
    }
}

// ---------------- main kernel: 8 queries x 4 corners per WG ----------------
__global__ __launch_bounds__(256) void liif_main(
        const float* __restrict__ feat, const float* __restrict__ coord,
        const float* __restrict__ cell,
        const float* __restrict__ b0, const float* __restrict__ b1,
        const float* __restrict__ b2, const float* __restrict__ b3,
        const float* __restrict__ b4,
        const _Float16* __restrict__ wsW,
        float* __restrict__ out) {
    // Xbuf doubles as the odd hidden buffer (X is dead after layer 0).
    __shared__ _Float16 Xbuf[MROWS * XSTR];        // 39936 B
    __shared__ _Float16 Heven[MROWS * HSTR];       // 17920 B
    __shared__ int   ixA[MROWS], iyA[MROWS];
    __shared__ float areaA[MROWS];
    __shared__ float predL[MROWS][3];
    __shared__ float b4L[3];

    _Float16* Hodd = Xbuf;                          // aliased, stride HSTR

    const int tid = threadIdx.x;
    const int wg  = blockIdx.x;                     // B * QD / 8 workgroups
    const int bb    = wg / (QD / 8);
    const int qbase = (wg % (QD / 8)) * 8;

    // ---- per-row (query,corner) geometry ----
    if (tid < MROWS) {
        int r  = tid;
        int qi = r >> 2, c = r & 3;
        int q  = qbase + qi;
        float c0 = coord[((size_t)bb * QD + q) * 2 + 0];
        float c1 = coord[((size_t)bb * QD + q) * 2 + 1];
        float vx = (c & 2) ? 1.0f : -1.0f;
        float vy = (c & 1) ? 1.0f : -1.0f;
        const float rx = 1.0f / HF, ry = 1.0f / WF, lim = 1.0f - 1e-6f;
        float cx = fminf(fmaxf(c0 + vx * rx + 1e-6f, -lim), lim);
        float cy = fminf(fmaxf(c1 + vy * ry + 1e-6f, -lim), lim);
        int ix = (int)fminf(fmaxf(floorf((cx + 1.0f) * HF * 0.5f), 0.0f), (float)(HF - 1));
        int iy = (int)fminf(fmaxf(floorf((cy + 1.0f) * WF * 0.5f), 0.0f), (float)(WF - 1));
        ixA[r] = ix; iyA[r] = iy;
        float qx = -1.0f + (2.0f * ix + 1.0f) / HF;
        float qy = -1.0f + (2.0f * iy + 1.0f) / WF;
        float rel0 = (c0 - qx) * HF;
        float rel1 = (c1 - qy) * WF;
        areaA[r] = fabsf(rel0 * rel1) + 1e-9f;
        float cl0 = cell[((size_t)bb * QD + q) * 2 + 0] * HF;
        float cl1 = cell[((size_t)bb * QD + q) * 2 + 1] * WF;
        _Float16* xr = Xbuf + r * XSTR;
        xr[576] = (_Float16)rel0; xr[577] = (_Float16)rel1;
        xr[578] = (_Float16)cl0;  xr[579] = (_Float16)cl1;
        for (int z = 580; z < K0PAD; ++z) xr[z] = (_Float16)0.0f;
    }
    if (tid < 3) b4L[tid] = b4[tid];
    __syncthreads();

    // ---- gather 3x3 unfolded features (feat is L2-resident) ----
    // 8 threads per row; thread owns channels c = g*8 .. g*8+7.
    // 9 tap offsets + validity computed once, reused for all 8 channels;
    // full unroll makes c/di/dj compile-time.
    {
        int r = tid >> 3;
        int g = tid & 7;
        int ix = ixA[r], iy = iyA[r];
        int   offs[9];
        bool  ok[9];
#pragma unroll
        for (int t = 0; t < 9; ++t) {
            int di = t / 3 - 1, dj = t % 3 - 1;       // constants after unroll
            int x = ix + di, y = iy + dj;
            ok[t]   = ((unsigned)x < (unsigned)HF) & ((unsigned)y < (unsigned)WF);
            offs[t] = x * WF + y;
        }
        const float* fb = feat + (size_t)bb * CCH * HWSZ + (size_t)(g * 8) * HWSZ;
        _Float16* xr = Xbuf + r * XSTR + g * 8 * 9;
#pragma unroll
        for (int j = 0; j < 8; ++j) {
            const float* fc = fb + (size_t)j * HWSZ;
#pragma unroll
            for (int t = 0; t < 9; ++t) {
                float v = ok[t] ? fc[offs[t]] : 0.0f;
                xr[j * 9 + t] = (_Float16)v;
            }
        }
    }
    __syncthreads();

    const int lane = tid & 31;
    const int wid  = tid >> 5;

    // ---- 4 hidden WMMA layers (ReLU fused); Hodd aliases Xbuf ----
    wmma_layer(Xbuf,  XSTR, KT0, wsW,         b0, Heven, HSTR, lane, wid); __syncthreads();
    wmma_layer(Heven, HSTR, KT1, wsW + BASE1, b1, Hodd,  HSTR, lane, wid); __syncthreads();
    wmma_layer(Hodd,  HSTR, KT1, wsW + BASE2, b2, Heven, HSTR, lane, wid); __syncthreads();
    wmma_layer(Heven, HSTR, KT1, wsW + BASE3, b3, Hodd,  HSTR, lane, wid); __syncthreads();

    // ---- output layer 256 -> 3 via WMMA (N padded to 16); waves 0-1 only ----
    if (wid < 2) {
        const int laneHi = lane >> 4;
        const int m      = lane & 15;
        const int nn     = lane & 15;
        const int mt     = wid;                 // M-tile per wave
        const _Float16* ap = Hodd + (mt * 16 + m) * HSTR + laneHi * 16;
        const _Float16* wp = wsW + BASE4 + lane * 16;
        v8f acc = {};
#pragma unroll
        for (int kt = 0; kt < KT1; ++kt) {
            v8h alo = *(const v8h*)(ap + kt * 32);
            v8h ahi = *(const v8h*)(ap + kt * 32 + 8);
            v16h a  = __builtin_shufflevector(alo, ahi, 0,1,2,3,4,5,6,7,8,9,10,11,12,13,14,15);
            v16h b  = *(const v16h*)(wp + kt * 512);
            acc = __builtin_amdgcn_wmma_f32_16x16x32_f16(false, a, false, b, (short)0, acc, false, false);
        }
        if (nn < 3) {
            float bj = b4L[nn];
#pragma unroll
            for (int r = 0; r < 8; ++r)
                predL[mt * 16 + r + laneHi * 8][nn] = acc[r] + bj;
        }
    }
    __syncthreads();

    // ---- local ensemble (diagonal area swap) ----
    if (tid < 8) {
        int qi = tid, rb = qi * 4;
        float a0 = areaA[rb + 0], a1 = areaA[rb + 1];
        float a2 = areaA[rb + 2], a3 = areaA[rb + 3];
        float inv = 1.0f / (a0 + a1 + a2 + a3);
        float wgt0 = a3 * inv, wgt1 = a2 * inv, wgt2 = a1 * inv, wgt3 = a0 * inv;
        int q = qbase + qi;
        float* o = out + ((size_t)bb * QD + q) * 3;
#pragma unroll
        for (int j = 0; j < 3; ++j) {
            o[j] = predL[rb + 0][j] * wgt0 + predL[rb + 1][j] * wgt1 +
                   predL[rb + 2][j] * wgt2 + predL[rb + 3][j] * wgt3;
        }
    }
}

extern "C" void kernel_launch(void* const* d_in, const int* in_sizes, int n_in,
                              void* d_out, int out_size, void* d_ws, size_t ws_size,
                              hipStream_t stream) {
    const float* feat  = (const float*)d_in[0];
    const float* coord = (const float*)d_in[1];
    const float* cell  = (const float*)d_in[2];
    const float* w0 = (const float*)d_in[3];
    const float* b0 = (const float*)d_in[4];
    const float* w1 = (const float*)d_in[5];
    const float* b1 = (const float*)d_in[6];
    const float* w2 = (const float*)d_in[7];
    const float* b2 = (const float*)d_in[8];
    const float* w3 = (const float*)d_in[9];
    const float* b3 = (const float*)d_in[10];
    const float* w4 = (const float*)d_in[11];
    const float* b4 = (const float*)d_in[12];
    _Float16* wsW = (_Float16*)d_ws;          // needs TOTAL_W*2 ~= 713 KB scratch

    // repack all weights (incl. padded w4) into WMMA B-fragment order (f16)
    liif_prep_weights<<<(TOTAL_W + 255) / 256, 256, 0, stream>>>(w0, w1, w2, w3, w4, wsW);

    // B * QD / 8 workgroups, each does 8 queries x 4 corners (M = 32 rows)
    const int nwg = 2 * (QD / 8);
    liif_main<<<nwg, 256, 0, stream>>>(feat, coord, cell, b0, b1, b2, b3,
                                       b4, wsW, (float*)d_out);
}
